// LinearMultiHeadAttention_35459249996190
// MI455X (gfx1250) — matrix operand order repaired and checked by reading, not compiled, not run on hardware
//
#include <hip/hip_runtime.h>
#include <hip/hip_bf16.h>

#define BB 4
#define NN 16384
#define HH 8
#define CHUNKS 64          // chunks per batch in kv-accumulation kernel
#define ITERS 2            // NN / (128 * CHUNKS)

typedef __attribute__((ext_vector_type(16))) __bf16 bf16x16;
typedef __attribute__((ext_vector_type(8)))  float  f32x8;

__device__ __forceinline__ f32x8 wmma_bf16(bf16x16 a, bf16x16 b, f32x8 c) {
  // D = A x B + C, 16x16x32 bf16 -> f32
  return __builtin_amdgcn_wmma_f32_16x16x32_bf16(false, a, false, b, (short)0, c, false, false);
}

// A-fragment (16x32, bf16): lane holds row, k = (e&7) + 8*(lane>=16) + 16*(e>=8)
// -> elements 0..7 contiguous at p, elements 8..15 contiguous at p+16
__device__ __forceinline__ bf16x16 ldfragA(const __bf16* p) {
  union { uint4 u[2]; bf16x16 v; } t;
  t.u[0] = *(const uint4*)(p);
  t.u[1] = *(const uint4*)(p + 16);
  return t.v;
}
// B-fragment (32x16, bf16): lane holds col, k = (e&7) + 8*(lane>=16) + 16*(e>=8)
__device__ __forceinline__ bf16x16 ldfragB(const __bf16* p) {
  union { uint4 u[2]; bf16x16 v; } t;
  t.u[0] = *(const uint4*)(p);
  t.u[1] = *(const uint4*)(p + 16);
  return t.v;
}

// [16 x 64] tile of C = A(128x64) x B(64x64), K=64 in 2 WMMA k-steps.
// A row-major bf16 LDS (stride sA), B stored transposed BT[n][k] (stride sB).
// Wave `mt` computes rows 16*mt..16*mt+15 into acc[0..3] (4 n-tiles).
__device__ __forceinline__ void gemm_tile(const __bf16* A, int sA,
                                          const __bf16* BT, int sB,
                                          int mt, int lane, f32x8 acc[4]) {
  const int hl = lane >> 4, ll = lane & 15;
  const int arow = 16 * mt + ll;
#pragma unroll
  for (int ks = 0; ks < 2; ++ks) {
    bf16x16 a = ldfragA(A + arow * sA + 32 * ks + 8 * hl);
#pragma unroll
    for (int nt = 0; nt < 4; ++nt) {
      bf16x16 b = ldfragB(BT + (16 * nt + ll) * sB + 32 * ks + 8 * hl);
      acc[nt] = wmma_bf16(a, b, acc[nt]);
    }
  }
}

// ---------------------------------------------------------------------------
// Kernel A: per (b, chunk): K/V projection + layernorm + RoPE(K) + partial
//           kv_h = K_h^T V_h accumulated over 256 rows, written to part[].
// ---------------------------------------------------------------------------
__global__ __launch_bounds__(256, 1) void kv_proj_kernel(
    const float* __restrict__ x, const float* __restrict__ pos,
    const float* __restrict__ Wk, const float* __restrict__ Wv,
    float* __restrict__ part) {
  __shared__ __align__(16) __bf16 xb[128 * 72];
  __shared__ __align__(16) __bf16 wkT[64 * 72];
  __shared__ __align__(16) __bf16 wvT[64 * 72];
  __shared__ __align__(16) __bf16 khsT[64 * 136];   // K transposed: [d][row]
  __shared__ __align__(16) __bf16 vhsT[64 * 136];   // V transposed: [e][row]
  __shared__ float posl[256];

  const int chunk = blockIdx.x, b = blockIdx.y;
  const int t = threadIdx.x, w = t >> 5, lane = t & 31;
  const int hl = lane >> 4, ll = lane & 15;
  const int rowBase = chunk * (128 * ITERS);

  posl[t] = pos[(size_t)b * NN + rowBase + t];
  const float LNF = 0.2878231366242557f;           // ln(10000)/32
  const float if0 = __expf(-LNF * (float)ll);      // inv_freq for j = ll
  const float if1 = __expf(-LNF * (float)(16 + ll));

  for (int h = 0; h < HH; ++h) {
    __syncthreads();  // protect wkT/wvT + khsT/vhsT against previous head
#pragma unroll
    for (int i = 0; i < 16; ++i) {                 // load Wk_h, Wv_h transposed
      int idx = t + i * 256;
      int d = idx >> 6, j = idx & 63;
      wkT[j * 72 + d] = (__bf16)Wk[d * 512 + h * 64 + j];
      wvT[j * 72 + d] = (__bf16)Wv[d * 512 + h * 64 + j];
    }
    f32x8 kvacc[2] = {};
    for (int it = 0; it < ITERS; ++it) {
      __syncthreads();                             // weights ready / xb reuse
      const float4* xg = (const float4*)(x + ((size_t)b * NN + rowBase + it * 128) * 64);
#pragma unroll
      for (int i = 0; i < 8; ++i) {                // x tile -> bf16 LDS
        int i4 = t + i * 256;
        float4 v = xg[i4];
        __bf16* dst = &xb[(i4 >> 4) * 72 + (i4 & 15) * 4];
        dst[0] = (__bf16)v.x; dst[1] = (__bf16)v.y;
        dst[2] = (__bf16)v.z; dst[3] = (__bf16)v.w;
      }
      __syncthreads();

      // ---- K stage: gemm, layernorm on frags, rope, store transposed ----
      {
        f32x8 kf[4] = {};
        gemm_tile(xb, 72, wkT, 72, w, lane, kf);
#pragma unroll
        for (int r = 0; r < 8; ++r) {
          float v0 = kf[0][r], v1 = kf[1][r], v2 = kf[2][r], v3 = kf[3][r];
          float s1 = v0 + v1 + v2 + v3;
          float s2 = v0 * v0 + v1 * v1 + v2 * v2 + v3 * v3;
#pragma unroll
          for (int m = 1; m < 16; m <<= 1) {       // row reduce across 16 lanes
            s1 += __shfl_xor(s1, m, 32);
            s2 += __shfl_xor(s2, m, 32);
          }
          float mu = s1 * 0.015625f;
          float rs = rsqrtf(s2 * 0.015625f - mu * mu + 1e-5f);
          v0 = (v0 - mu) * rs; v1 = (v1 - mu) * rs;
          v2 = (v2 - mu) * rs; v3 = (v3 - mu) * rs;
          int mloc = 16 * w + r + 8 * hl;
          float tp = posl[it * 128 + mloc] * 64.0f;
          float s0, c0, s1r, c1r;
          __sincosf(tp * if0, &s0, &c0);
          __sincosf(tp * if1, &s1r, &c1r);
          kf[0][r] = v0 * c0 - v2 * s0;            // d and d+32 partners are
          kf[1][r] = v1 * c1r - v3 * s1r;          // tiles (0,2) and (1,3) at
          kf[2][r] = v2 * c0 + v0 * s0;            // the same lane/VGPR slot
          kf[3][r] = v3 * c1r + v1 * s1r;
        }
#pragma unroll
        for (int ti = 0; ti < 4; ++ti)
#pragma unroll
          for (int r = 0; r < 8; ++r)
            khsT[(16 * ti + ll) * 136 + 16 * w + r + 8 * hl] = (__bf16)kf[ti][r];
      }
      // ---- V stage: gemm, layernorm, store transposed ----
      {
        f32x8 vf[4] = {};
        gemm_tile(xb, 72, wvT, 72, w, lane, vf);
#pragma unroll
        for (int r = 0; r < 8; ++r) {
          float v0 = vf[0][r], v1 = vf[1][r], v2 = vf[2][r], v3 = vf[3][r];
          float s1 = v0 + v1 + v2 + v3;
          float s2 = v0 * v0 + v1 * v1 + v2 * v2 + v3 * v3;
#pragma unroll
          for (int m = 1; m < 16; m <<= 1) {
            s1 += __shfl_xor(s1, m, 32);
            s2 += __shfl_xor(s2, m, 32);
          }
          float mu = s1 * 0.015625f;
          float rs = rsqrtf(s2 * 0.015625f - mu * mu + 1e-5f);
          vf[0][r] = (v0 - mu) * rs; vf[1][r] = (v1 - mu) * rs;
          vf[2][r] = (v2 - mu) * rs; vf[3][r] = (v3 - mu) * rs;
        }
#pragma unroll
        for (int ti = 0; ti < 4; ++ti)
#pragma unroll
          for (int r = 0; r < 8; ++r)
            vhsT[(16 * ti + ll) * 136 + 16 * w + r + 8 * hl] = (__bf16)vf[ti][r];
      }
      __syncthreads();
      // ---- kv partial: kv[64,64] += K^T V, K=128 in 4 k-steps ----
#pragma unroll
      for (int i = 0; i < 2; ++i) {
        int tile = w * 2 + i;
        int mt = tile >> 2, nt = tile & 3;
#pragma unroll
        for (int ks = 0; ks < 4; ++ks) {
          bf16x16 a = ldfragA(khsT + (16 * mt + ll) * 136 + 32 * ks + 8 * hl);
          bf16x16 bb = ldfragB(vhsT + (16 * nt + ll) * 136 + 32 * ks + 8 * hl);
          kvacc[i] = wmma_bf16(a, bb, kvacc[i]);
        }
      }
    }
    // deterministic partial writes (no float atomics)
    float* pp = part + ((((size_t)b * CHUNKS + chunk) * HH + h) << 12);
#pragma unroll
    for (int i = 0; i < 2; ++i) {
      int tile = w * 2 + i;
      int mt = tile >> 2, nt = tile & 3;
#pragma unroll
      for (int r = 0; r < 8; ++r) {
        int d = 16 * mt + r + 8 * hl, e = 16 * nt + ll;
        pp[d * 64 + e] = kvacc[i][r];
      }
    }
  }
}

// ---------------------------------------------------------------------------
// Reduce: kv[b][h][64][64] = sum over CHUNKS partials (deterministic order)
// ---------------------------------------------------------------------------
__global__ __launch_bounds__(256, 1) void kv_reduce_kernel(
    const float* __restrict__ part, float* __restrict__ kvg) {
  int idx = blockIdx.x * 256 + threadIdx.x;        // 131072 elements
  int b = idx >> 15;
  int rem = idx & 32767;
  const float* p = part + ((size_t)b * CHUNKS) * (HH * 4096) + rem;
  float s = 0.f;
  for (int c = 0; c < CHUNKS; ++c) s += p[(size_t)c * (HH * 4096)];
  kvg[idx] = s;
}

// ---------------------------------------------------------------------------
// Kernel B: Q projection + RoPE, z = Q @ (kv/N), out += z @ Wo_h over heads
// ---------------------------------------------------------------------------
__global__ __launch_bounds__(256, 1) void out_proj_kernel(
    const float* __restrict__ x, const float* __restrict__ pos,
    const float* __restrict__ Wq, const float* __restrict__ kvg,
    const float* __restrict__ Wo, float* __restrict__ out) {
  __shared__ __align__(16) __bf16 xb[128 * 72];
  __shared__ __align__(16) __bf16 wqT[64 * 72];
  __shared__ __align__(16) __bf16 qhs[128 * 72];
  __shared__ __align__(16) __bf16 kvbT[64 * 72];
  __shared__ __align__(16) __bf16 zs[128 * 72];
  __shared__ __align__(16) __bf16 woT[64 * 72];
  __shared__ float posl[128];

  const int nb = blockIdx.x, b = blockIdx.y;
  const int t = threadIdx.x, w = t >> 5, lane = t & 31;
  const int hl = lane >> 4, ll = lane & 15;
  const int rowBase = nb * 128;

  if (t < 128) posl[t] = pos[(size_t)b * NN + rowBase + t];
  {
    const float4* xg = (const float4*)(x + ((size_t)b * NN + rowBase) * 64);
#pragma unroll
    for (int i = 0; i < 8; ++i) {
      int i4 = t + i * 256;
      float4 v = xg[i4];
      __bf16* dst = &xb[(i4 >> 4) * 72 + (i4 & 15) * 4];
      dst[0] = (__bf16)v.x; dst[1] = (__bf16)v.y;
      dst[2] = (__bf16)v.z; dst[3] = (__bf16)v.w;
    }
  }
  const float LNF = 0.2878231366242557f;
  const float if0 = __expf(-LNF * (float)ll);
  const float if1 = __expf(-LNF * (float)(16 + ll));
  const float invN = 1.0f / (float)NN;

  f32x8 oacc[4] = {};                               // persistent across heads
  for (int h = 0; h < HH; ++h) {
    __syncthreads();
#pragma unroll
    for (int i = 0; i < 16; ++i) {                  // Wq_h, Wo_h, kv_h -> LDS
      int idx = t + i * 256;
      int r0 = idx >> 6, c0 = idx & 63;
      wqT[c0 * 72 + r0] = (__bf16)Wq[r0 * 512 + h * 64 + c0];
      woT[c0 * 72 + r0] = (__bf16)Wo[((size_t)h * 64 + r0) * 64 + c0];
      kvbT[c0 * 72 + r0] = (__bf16)(kvg[(((size_t)b * HH + h) << 12) + idx] * invN);
    }
    __syncthreads();
    // ---- Q = x @ Wq_h, RoPE on frags ----
    f32x8 qf[4] = {};
    gemm_tile(xb, 72, wqT, 72, w, lane, qf);
#pragma unroll
    for (int r = 0; r < 8; ++r) {
      int mloc = 16 * w + r + 8 * hl;
      float tp = posl[mloc] * 64.0f;
      float s0, c0, s1r, c1r;
      __sincosf(tp * if0, &s0, &c0);
      __sincosf(tp * if1, &s1r, &c1r);
      float q0 = qf[0][r], q1 = qf[1][r], q2 = qf[2][r], q3 = qf[3][r];
      qf[0][r] = q0 * c0 - q2 * s0;
      qf[1][r] = q1 * c1r - q3 * s1r;
      qf[2][r] = q2 * c0 + q0 * s0;
      qf[3][r] = q3 * c1r + q1 * s1r;
    }
#pragma unroll
    for (int ti = 0; ti < 4; ++ti)
#pragma unroll
      for (int r = 0; r < 8; ++r)
        qhs[(16 * w + r + 8 * hl) * 72 + 16 * ti + ll] = (__bf16)qf[ti][r];
    __syncthreads();
    // ---- z = Q @ (kv/N) ----
    f32x8 zf[4] = {};
    gemm_tile(qhs, 72, kvbT, 72, w, lane, zf);
#pragma unroll
    for (int ti = 0; ti < 4; ++ti)
#pragma unroll
      for (int r = 0; r < 8; ++r)
        zs[(16 * w + r + 8 * hl) * 72 + 16 * ti + ll] = (__bf16)zf[ti][r];
    __syncthreads();
    // ---- out += z @ Wo_h ----
    gemm_tile(zs, 72, woT, 72, w, lane, oacc);
  }
#pragma unroll
  for (int ti = 0; ti < 4; ++ti)
#pragma unroll
    for (int r = 0; r < 8; ++r)
      out[((size_t)b * NN + rowBase + 16 * w + r + 8 * hl) * 64 + 16 * ti + ll] =
          oacc[ti][r];
}

// ---------------------------------------------------------------------------
extern "C" void kernel_launch(void* const* d_in, const int* in_sizes, int n_in,
                              void* d_out, int out_size, void* d_ws, size_t ws_size,
                              hipStream_t stream) {
  const float* x   = (const float*)d_in[0];
  const float* pos = (const float*)d_in[1];
  const float* Wq  = (const float*)d_in[2];
  const float* Wk  = (const float*)d_in[3];
  const float* Wv  = (const float*)d_in[4];
  const float* Wo  = (const float*)d_in[5];
  float* out = (float*)d_out;

  // ws layout: part[B][CHUNKS][H][64][64] f32 (32 MB), kv[B][H][64][64] f32 (512 KB)
  float* part = (float*)d_ws;
  float* kvg  = part + (size_t)BB * CHUNKS * HH * 4096;

  kv_proj_kernel<<<dim3(CHUNKS, BB), 256, 0, stream>>>(x, pos, Wk, Wv, part);
  kv_reduce_kernel<<<dim3((BB * HH * 4096) / 256), 256, 0, stream>>>(part, kvg);
  out_proj_kernel<<<dim3(NN / 128, BB), 256, 0, stream>>>(x, pos, Wq, kvg, Wo, out);
}
